// TemporalAttention_42975442764284
// MI455X (gfx1250) — matrix operand order repaired and checked by reading, not compiled, not run on hardware
//
#include <hip/hip_runtime.h>

// Problem constants (from reference): b=8, c=256, c_qk=32, n=64*64=4096
#define B_   8
#define C_   256
#define CQK_ 32
#define N_   4096
#define LOG2E_ 1.4426950408889634f

typedef __attribute__((ext_vector_type(16))) __bf16 v16bf;
typedef __attribute__((ext_vector_type(8)))  float  v8f;

union BF16Tile { v16bf v; uint4 q[2]; };

// ---------------------------------------------------------------------------
// Kernel 1: QKV projection (1x1 convs), fp32 accumulate, bf16 output.
//   Qb: [b][n][32]  pre-scaled by log2(e)  -> WMMA B-tiles for K.Q^T
//   Kb: [b][n][32]                         -> WMMA A-tiles for K.Q^T
//   Vb: [b][256][n] channel-major          -> WMMA A-tiles for V.P^T
// Pre-scaling Q by log2e makes scores come out of the WMMA in base-2 units,
// so softmax needs only raw v_exp_f32 (exp2), no per-element log2e multiply.
// ---------------------------------------------------------------------------
__global__ __launch_bounds__(256)
void qkv_proj_kernel(const float* __restrict__ x,
                     const float* __restrict__ wq, const float* __restrict__ bq,
                     const float* __restrict__ wk, const float* __restrict__ bk,
                     const float* __restrict__ wv, const float* __restrict__ bv,
                     __bf16* __restrict__ Qb, __bf16* __restrict__ Kb,
                     __bf16* __restrict__ Vb)
{
    __shared__ float xs[C_][8];
    const int p0 = blockIdx.x * 8;
    const int b  = blockIdx.y;
    const int t  = threadIdx.x;

    // Stage x tile [256 channels x 8 pixels] into LDS (two b128 per thread).
    {
        const float* src = x + ((size_t)b * C_ + t) * N_ + p0;
        float4 a = *(const float4*)src;
        float4 c = *(const float4*)(src + 4);
        *(float4*)&xs[t][0] = a;
        *(float4*)&xs[t][4] = c;
    }
    __syncthreads();

    // V projection: thread t -> output channel t, 8 pixels.
    {
        float acc[8] = {0.f,0.f,0.f,0.f,0.f,0.f,0.f,0.f};
        const float* wrow = wv + (size_t)t * C_;
        for (int c = 0; c < C_; ++c) {
            const float w = wrow[c];
#pragma unroll
            for (int p = 0; p < 8; ++p) acc[p] = fmaf(w, xs[c][p], acc[p]);
        }
        const float bias = bv[t];
        union { uint4 q; __bf16 h[8]; } ov;
#pragma unroll
        for (int p = 0; p < 8; ++p) ov.h[p] = (__bf16)(acc[p] + bias);
        *(uint4*)(Vb + ((size_t)b * C_ + t) * N_ + p0) = ov.q;   // one 16B store
    }

    // Q/K projections: threads 0..31 -> Q channel t, 32..63 -> K channel t-32.
    if (t < 64) {
        const bool isq = (t < 32);
        const int  o   = isq ? t : (t - 32);
        const float* wrow = (isq ? wq : wk) + (size_t)o * C_;
        float acc[8] = {0.f,0.f,0.f,0.f,0.f,0.f,0.f,0.f};
        for (int c = 0; c < C_; ++c) {
            const float w = wrow[c];
#pragma unroll
            for (int p = 0; p < 8; ++p) acc[p] = fmaf(w, xs[c][p], acc[p]);
        }
        const float bias = isq ? bq[o] : bk[o];
        const float sc   = isq ? LOG2E_ : 1.0f;   // fold log2e into Q
        __bf16* dst = isq ? Qb : Kb;
#pragma unroll
        for (int p = 0; p < 8; ++p)
            dst[((size_t)b * N_ + p0 + p) * CQK_ + o] = (__bf16)((acc[p] + bias) * sc);
    }
}

// ---------------------------------------------------------------------------
// Kernel 2: flash attention, bf16 WMMA, fp32 accumulate, base-2 online softmax.
// TRANSPOSED dataflow: S^T = K.Q^T and out^T = V.P^T -> query is a LANE in
// every C/D tile; softmax state is per-lane scalar (one xor-16 shuffle/step).
// 64 keys per step: 4 score WMMAs + 8 PV WMMAs, fixed per-step overhead
// (acc rescale, shuffle+wait, addressing) amortized over 2x matrix work.
// Grid: (N/16 query tiles, B). Block: 128 threads = 4 waves.
// ---------------------------------------------------------------------------
__global__ __launch_bounds__(128)
void flash_attn_kernel(const __bf16* __restrict__ Qb,
                       const __bf16* __restrict__ Kb,
                       const __bf16* __restrict__ Vb,
                       float* __restrict__ out)
{
    const int i0   = blockIdx.x * 16;      // query tile base (pixel index)
    const int b    = blockIdx.y;
    const int wave = threadIdx.x >> 5;     // 0..3
    const int lane = threadIdx.x & 31;
    const int half = lane >> 4;
    const int l16  = lane & 15;

    // Per-wave staging for P^T: 16 query rows x 72 bf16 (144B stride; every
    // 16B chunk used below stays 16B-aligned).
    __shared__ __bf16 pstage[4][16 * 72];
    __bf16* mystage = &pstage[wave][0];

    // --- Q tile in 16-bit B layout (32K x 16N): lane = query column l16;
    // K dims {d0..d0+7, d0+16..d0+23}, d0 = half*8. Loaded once.
    BF16Tile qb;
    {
        const __bf16* qrow = Qb + ((size_t)b * N_ + (i0 + l16)) * CQK_ + half * 8;
        qb.q[0] = *(const uint4*)qrow;
        qb.q[1] = *(const uint4*)(qrow + 16);
    }

    v8f acc[4] = {};          // out^T accumulators: slot = channel, lane = query
    float m_run = -3.0e38f;   // per-lane (query) running max (base-2 units)
    float l_run = 0.f;        // per-lane partial softmax sum (this half's keys)

    for (int j0 = 0; j0 < N_; j0 += 64) {
        // --- K A-tiles (16M x 32K): tile t rows = keys j0 + 16t + l16;
        // lane dims K = {d0..d0+7, d0+16..d0+23}, d0 = half*8.
        BF16Tile ka[4];
        {
            const __bf16* kp = Kb + ((size_t)b * N_ + j0 + l16) * CQK_ + half * 8;
#pragma unroll
            for (int t = 0; t < 4; ++t) {
                ka[t].q[0] = *(const uint4*)(kp + t * 16 * CQK_);
                ka[t].q[1] = *(const uint4*)(kp + t * 16 * CQK_ + 16);
            }
            if (j0 + 64 < N_)
                __builtin_prefetch(kp + 64 * CQK_, 0, 3);
        }

        // --- S^T tiles [keys x queries]: s[t] slot r = key j0+16t+(half?8:0)+r.
        v8f s[4];
#pragma unroll
        for (int t = 0; t < 4; ++t) {
            v8f z = {};
            s[t] = __builtin_amdgcn_wmma_f32_16x16x32_bf16(false, ka[t].v, false, qb.v,
                                                           (short)0, z, false, false);
        }

        // --- per-query max: in-lane tree over 32 slots + one xor-16 combine.
        float v = fmaxf(fmaxf(s[0][0], s[1][0]), fmaxf(s[2][0], s[3][0]));
#pragma unroll
        for (int r = 1; r < 8; ++r)
            v = fmaxf(v, fmaxf(fmaxf(s[0][r], s[1][r]), fmaxf(s[2][r], s[3][r])));
        v = fmaxf(v, __shfl_xor(v, 16));
        const float mnew  = fmaxf(m_run, v);
        const float scale = __builtin_amdgcn_exp2f(m_run - mnew);
        m_run = mnew;

        // --- exp2, per-lane sum, pack P^T into LDS (four 16B stores).
        float rs = 0.f;
#pragma unroll
        for (int t = 0; t < 4; ++t) {
            union { uint4 q; __bf16 h[8]; } pk;
#pragma unroll
            for (int r = 0; r < 8; ++r) {
                const float e = __builtin_amdgcn_exp2f(s[t][r] - m_run);
                rs += e;
                pk.h[r] = (__bf16)e;
            }
            // row = query l16; tile t cols 16t + half*8 + r (contiguous).
            *(uint4*)(mystage + l16 * 72 + t * 16 + half * 8) = pk.q;
        }
        l_run = l_run * scale + rs;

        // --- rescale accumulators by per-lane scalar (no broadcast needed).
#pragma unroll
        for (int tt = 0; tt < 4; ++tt)
#pragma unroll
            for (int r = 0; r < 8; ++r) acc[tt][r] *= scale;

        // LDS stores -> loads within this wave (in-order DS, but be explicit).
        asm volatile("s_wait_dscnt 0x0" ::: "memory");

        // --- P^T B-tiles (32K x 16N): lane = query col l16;
        // pb0 = keys j0..j0+31 (cols half*8, +16), pb1 = keys j0+32..j0+63 (+32).
        BF16Tile pb0, pb1;
        {
            const __bf16* prow = mystage + l16 * 72 + half * 8;
            pb0.q[0] = *(const uint4*)prow;
            pb0.q[1] = *(const uint4*)(prow + 16);
            pb1.q[0] = *(const uint4*)(prow + 32);
            pb1.q[1] = *(const uint4*)(prow + 48);
        }

        // --- V A-tiles (16M x 32K): row M = channel, K = keys contiguous in Vb[c][n].
        const __bf16* vbase =
            Vb + ((size_t)(b * C_ + wave * 64 + l16)) * N_ + j0 + half * 8;
#pragma unroll
        for (int tt = 0; tt < 4; ++tt) {
            const __bf16* vp = vbase + (size_t)(tt * 16) * N_;
            BF16Tile va0, va1;
            va0.q[0] = *(const uint4*)vp;
            va0.q[1] = *(const uint4*)(vp + 16);
            va1.q[0] = *(const uint4*)(vp + 32);
            va1.q[1] = *(const uint4*)(vp + 48);
            acc[tt] = __builtin_amdgcn_wmma_f32_16x16x32_bf16(
                false, va0.v, false, pb0.v, (short)0, acc[tt], false, false);
            acc[tt] = __builtin_amdgcn_wmma_f32_16x16x32_bf16(
                false, va1.v, false, pb1.v, (short)0, acc[tt], false, false);
        }
    }

    // --- finalize: combine the two halves' partial sums, normalize, store.
    const float l    = l_run + __shfl_xor(l_run, 16);
    const float rinv = 1.0f / l;

    // out^T D layout: slot r -> channel (half? r+8 : r), lane -> query i0+l16.
#pragma unroll
    for (int tt = 0; tt < 4; ++tt) {
#pragma unroll
        for (int r = 0; r < 8; ++r) {
            const int ch = wave * 64 + tt * 16 + (half ? (r + 8) : r);
            out[((size_t)(b * C_ + ch)) * N_ + i0 + l16] = acc[tt][r] * rinv;
        }
    }
}

// ---------------------------------------------------------------------------
extern "C" void kernel_launch(void* const* d_in, const int* in_sizes, int n_in,
                              void* d_out, int out_size, void* d_ws, size_t ws_size,
                              hipStream_t stream)
{
    (void)in_sizes; (void)n_in; (void)out_size; (void)ws_size;

    const float* x  = (const float*)d_in[0];
    const float* wq = (const float*)d_in[1];
    const float* bq = (const float*)d_in[2];
    const float* wk = (const float*)d_in[3];
    const float* bk = (const float*)d_in[4];
    const float* wv = (const float*)d_in[5];
    const float* bv = (const float*)d_in[6];
    float* out = (float*)d_out;

    // Workspace layout (bf16): Q 2MB | K 2MB | V 16MB  => 20 MB total.
    __bf16* Qb = (__bf16*)d_ws;
    __bf16* Kb = Qb + (size_t)B_ * N_ * CQK_;
    __bf16* Vb = Kb + (size_t)B_ * N_ * CQK_;

    qkv_proj_kernel<<<dim3(N_ / 8, B_), 256, 0, stream>>>(
        x, wq, bq, wk, bk, wv, bv, Qb, Kb, Vb);
    flash_attn_kernel<<<dim3(N_ / 16, B_), 128, 0, stream>>>(Qb, Kb, Vb, out);
}
